// Model_80530636800154
// MI455X (gfx1250) — matrix operand order repaired and checked by reading, not compiled, not run on hardware
//
#include <hip/hip_runtime.h>

// ---------------------------------------------------------------------------
// 2-layer LSTM (H=256, B=512, T=256, I=2) for MI455X / gfx1250.
// Persistent kernel: 32 blocks x 16 batch rows, 8 wave32s per block.
// Recurrent GEMMs done with V_WMMA_F32_16X16X32_BF16 (f32 accumulate).
// Weights pre-packed to bf16 fragment layout, streamed from L2 every step
// (1.5 MB total -> fully L2 resident; pointer laundering stops LICM/spills).
// ---------------------------------------------------------------------------

#define HP      264      // padded LDS row stride (bf16 elems) -> no bank conflicts
#define TSTEPS  256
#define NB      16       // batch rows per block

typedef __attribute__((ext_vector_type(16))) __bf16 v16bf;
typedef __attribute__((ext_vector_type(8)))  float  v8f;
typedef __attribute__((ext_vector_type(4)))  int    i128;

union FragU { i128 q[2]; v16bf v; };

__device__ __forceinline__ unsigned short f32_to_bf16(float f) {
    unsigned u = __builtin_bit_cast(unsigned, f);
    u += 0x7FFFu + ((u >> 16) & 1u);            // round-to-nearest-even
    return (unsigned short)(u >> 16);
}
__device__ __forceinline__ float bf16_to_f32(unsigned short h) {
    return __builtin_bit_cast(float, (unsigned)h << 16);
}
__device__ __forceinline__ float sigmoid_f(float x) { return 1.f / (1.f + __expf(-x)); }
__device__ __forceinline__ float tanh_f(float x) {
    float e = __expf(-2.f * x);
    return (1.f - e) / (1.f + e);
}

// Defeat LICM: make the weight base pointer look loop-variant so the 192
// fragment loads stay inside the timestep loop (L2 streaming, no spills).
__device__ __forceinline__ const unsigned short* launder(const unsigned short* p) {
    unsigned long long v = (unsigned long long)p;
    asm volatile("" : "+s"(v));
    return (const unsigned short*)v;
}

// ---------------------------------------------------------------------------
// Weight pre-pack: W (1024x256 f32, row-major W[n][k]) -> bf16 fragments laid
// out so each lane's 16 B-elements are 32 contiguous bytes.
//   flat = ((((w*8 + kb)*8 + tt)*32 + lane)*16 + e)
//   n = (tt>>1)*256 + w*32 + (tt&1)*16 + (lane&15)          (gate-aligned tiles)
//   k = kb*32 + (lane>>4)*16 + e                            (contiguous k per lane half, same as A)
// ---------------------------------------------------------------------------
__global__ __launch_bounds__(256) void pack_weights(
    const float* __restrict__ Whh1, const float* __restrict__ Wih2,
    const float* __restrict__ Whh2, unsigned short* __restrict__ Wp)
{
    int flat = blockIdx.x * 256 + threadIdx.x;
    if (flat >= 3 * 262144) return;
    int mat  = flat >> 18;            // 262144 = 2^18 elements per matrix
    int rem  = flat & 0x3FFFF;
    int e    = rem & 15;
    int lane = (rem >> 4) & 31;
    int tt   = (rem >> 9) & 7;
    int kb   = (rem >> 12) & 7;
    int w    = (rem >> 15) & 7;
    int n = (tt >> 1) * 256 + w * 32 + (tt & 1) * 16 + (lane & 15);
    int k = kb * 32 + (lane >> 4) * 16 + e;
    const float* src = (mat == 0) ? Whh1 : (mat == 1) ? Wih2 : Whh2;
    Wp[flat] = f32_to_bf16(src[n * 256 + k]);
}

// 8 N-tiles x 8 K-blocks of V_WMMA_F32_16X16X32_BF16.
// Per K-block: issue A (2x ds_load_b128) + all 8 B tiles (16x global_load_b128)
// as one clause, then run the 8 WMMAs against them.
__device__ __forceinline__ void gemm_8x8(const unsigned short* hs,
                                         const unsigned short* ws,
                                         v8f* acc, int n, int hh, int lane)
{
#pragma unroll
    for (int kb = 0; kb < 8; ++kb) {
        FragU a;  // A 16x32 bf16: lane half hh -> K = hh*16+{0..15}
        const i128* ap = (const i128*)(hs + n * HP + kb * 32 + hh * 16);
        a.q[0] = ap[0];
        a.q[1] = ap[1];
        FragU b[8];
#pragma unroll
        for (int tt = 0; tt < 8; ++tt) {
            const i128* bp = (const i128*)(ws + (size_t)((kb * 8 + tt) * 32 + lane) * 16);
            b[tt].q[0] = bp[0];
            b[tt].q[1] = bp[1];
        }
#pragma unroll
        for (int tt = 0; tt < 8; ++tt)
            acc[tt] = __builtin_amdgcn_wmma_f32_16x16x32_bf16(
                false, a.v, false, b[tt].v, (short)0, acc[tt], false, false);
    }
}

__global__ __launch_bounds__(256) void lstm_persistent(
    const float* __restrict__ inputs,
    const float* __restrict__ W_ih1,
    const float* __restrict__ b_ih1, const float* __restrict__ b_hh1,
    const float* __restrict__ b_ih2, const float* __restrict__ b_hh2,
    const float* __restrict__ W_lin, const float* __restrict__ b_lin,
    const unsigned short* __restrict__ Wp,
    float* __restrict__ out)
{
    __shared__ __align__(16) unsigned short h1s[NB * HP];
    __shared__ __align__(16) unsigned short h2s[NB * HP];
    __shared__ float x_s[32];
    __shared__ float y_s[32];

    const int tid  = threadIdx.x;
    const int lane = tid & 31;
    const int w    = tid >> 5;        // wave id 0..7 (wave32)
    const int n    = lane & 15;       // N column within tile / batch row for A / y
    const int hh   = lane >> 4;       // lane half
    const int wg   = blockIdx.x;      // batch tile: rows [wg*16, wg*16+16)

    for (int i = tid; i < NB * HP; i += 256) { h1s[i] = 0; h2s[i] = 0; }

    // Per-tile constants: gate-aligned columns col = gate*256 + w*32 + tsub*16 + n
    float bias1[8], bias2[8], wi0[8], wi1[8];
#pragma unroll
    for (int tt = 0; tt < 8; ++tt) {
        int col   = (tt >> 1) * 256 + w * 32 + (tt & 1) * 16 + n;
        bias1[tt] = b_ih1[col] + b_hh1[col];
        bias2[tt] = b_ih2[col] + b_hh2[col];
        wi0[tt]   = W_ih1[col * 2 + 0];
        wi1[tt]   = W_ih1[col * 2 + 1];
    }
    // Output projection slice: lane = o*16 + m, o = hh, m = n
    float wl[32];
#pragma unroll
    for (int jj = 0; jj < 32; ++jj) wl[jj] = W_lin[hh * 256 + w * 32 + jj];
    const float blin = b_lin[hh];

    float c1[2][8] = {}, c2[2][8] = {};   // wave's slice of cell states, in regs

    __syncthreads();

#pragma unroll 1
    for (int t = 0; t < TSTEPS; ++t) {
        // Re-derive weight slice pointers each step through an opaque asm so
        // the fragment loads cannot be hoisted out of the loop and spilled.
        const unsigned short* Whh1p = launder(Wp) + (size_t)w * 32768;
        const unsigned short* Wih2p = launder(Wp + 262144) + (size_t)w * 32768;
        const unsigned short* Whh2p = launder(Wp + 2 * 262144) + (size_t)w * 32768;

        if (w == 0) {
            // stage x_t (16 rows x 2) and reset y accumulator to bias
            x_s[lane] = inputs[(((size_t)(wg * NB + (lane >> 1))) * TSTEPS + t) * 2 + (lane & 1)];
            y_s[lane] = blin;
        }
        __syncthreads();

        // ---- cell 1: gates = bias + x@W_ih1^T + h1@W_hh1^T ----
        float xm0[8], xm1[8];
#pragma unroll
        for (int r = 0; r < 8; ++r) { int mm = r + 8 * hh; xm0[r] = x_s[mm * 2]; xm1[r] = x_s[mm * 2 + 1]; }

        v8f acc[8];
#pragma unroll
        for (int tt = 0; tt < 8; ++tt)
#pragma unroll
            for (int r = 0; r < 8; ++r)
                acc[tt][r] = bias1[tt] + wi0[tt] * xm0[r] + wi1[tt] * xm1[r];

        gemm_8x8(h1s, Whh1p, acc, n, hh, lane);

        float h1n[2][8];
#pragma unroll
        for (int t2 = 0; t2 < 2; ++t2)
#pragma unroll
            for (int r = 0; r < 8; ++r) {
                float ig = acc[0 + t2][r];   // tiles 0,1 = gate i
                float fg = acc[2 + t2][r];   // tiles 2,3 = gate f
                float gg = acc[4 + t2][r];   // tiles 4,5 = gate g
                float og = acc[6 + t2][r];   // tiles 6,7 = gate o
                float cn = sigmoid_f(fg) * c1[t2][r] + sigmoid_f(ig) * tanh_f(gg);
                c1[t2][r] = cn;
                h1n[t2][r] = sigmoid_f(og) * tanh_f(cn);
            }

        __syncthreads();                     // everyone done reading old h1
#pragma unroll
        for (int t2 = 0; t2 < 2; ++t2) {
            int j = w * 32 + t2 * 16 + n;
#pragma unroll
            for (int r = 0; r < 8; ++r)
                h1s[(r + 8 * hh) * HP + j] = f32_to_bf16(h1n[t2][r]);
        }
        __syncthreads();                     // h1_new visible

        // ---- cell 2: gates = bias + h1_new@W_ih2^T + h2@W_hh2^T ----
#pragma unroll
        for (int tt = 0; tt < 8; ++tt)
#pragma unroll
            for (int r = 0; r < 8; ++r)
                acc[tt][r] = bias2[tt];

        gemm_8x8(h1s, Wih2p, acc, n, hh, lane);
        gemm_8x8(h2s, Whh2p, acc, n, hh, lane);

        float h2n[2][8];
#pragma unroll
        for (int t2 = 0; t2 < 2; ++t2)
#pragma unroll
            for (int r = 0; r < 8; ++r) {
                float ig = acc[0 + t2][r];
                float fg = acc[2 + t2][r];
                float gg = acc[4 + t2][r];
                float og = acc[6 + t2][r];
                float cn = sigmoid_f(fg) * c2[t2][r] + sigmoid_f(ig) * tanh_f(gg);
                c2[t2][r] = cn;
                h2n[t2][r] = sigmoid_f(og) * tanh_f(cn);
            }

        __syncthreads();                     // everyone done reading old h2
#pragma unroll
        for (int t2 = 0; t2 < 2; ++t2) {
            int j = w * 32 + t2 * 16 + n;
#pragma unroll
            for (int r = 0; r < 8; ++r)
                h2s[(r + 8 * hh) * HP + j] = f32_to_bf16(h2n[t2][r]);
        }
        __syncthreads();                     // h2_new visible

        // ---- y = h2_new @ W_lin^T + b_lin : each wave reduces its 32 j's ----
        float part = 0.f;
#pragma unroll
        for (int jj = 0; jj < 32; ++jj)
            part = fmaf(bf16_to_f32(h2s[n * HP + w * 32 + jj]), wl[jj], part);
        atomicAdd(&y_s[hh * 16 + n], part);  // ds_add_f32
        __syncthreads();

        if (w == 0)
            out[(((size_t)(wg * NB + n)) * TSTEPS + t) * 2 + hh] = y_s[lane];
    }
}

extern "C" void kernel_launch(void* const* d_in, const int* in_sizes, int n_in,
                              void* d_out, int out_size, void* d_ws, size_t ws_size,
                              hipStream_t stream)
{
    const float* inputs = (const float*)d_in[0];
    const float* W_ih1  = (const float*)d_in[1];
    const float* W_hh1  = (const float*)d_in[2];
    const float* b_ih1  = (const float*)d_in[3];
    const float* b_hh1  = (const float*)d_in[4];
    const float* W_ih2  = (const float*)d_in[5];
    const float* W_hh2  = (const float*)d_in[6];
    const float* b_ih2  = (const float*)d_in[7];
    const float* b_hh2  = (const float*)d_in[8];
    const float* W_lin  = (const float*)d_in[9];
    const float* b_lin  = (const float*)d_in[10];

    unsigned short* Wp = (unsigned short*)d_ws;   // 3 * 262144 bf16 = 1.5 MB
    float* out = (float*)d_out;

    pack_weights<<<dim3((3 * 262144 + 255) / 256), dim3(256), 0, stream>>>(
        W_hh1, W_ih2, W_hh2, Wp);
    lstm_persistent<<<dim3(32), dim3(256), 0, stream>>>(
        inputs, W_ih1, b_ih1, b_hh1, b_ih2, b_hh2, W_lin, b_lin, Wp, out);
}
